// DLASSO_GNNHyp2_10677288698535
// MI455X (gfx1250) — matrix operand order repaired and checked by reading, not compiled, not run on hardware
//
#include <hip/hip_runtime.h>
#include <hip/hip_bf16.h>

#define B_  8
#define P_  20
#define M_  512
#define N_  1024
#define H_  128
#define K_  5
#define F_  (2*N_)     // 2048
#define H4_ (4*H_)     // 512
#define NEGS 0.01f
#define EPSV 1e-5f

typedef __attribute__((ext_vector_type(16))) __bf16 v16bf;
typedef __attribute__((ext_vector_type(8)))  __bf16 v8bf;
typedef __attribute__((ext_vector_type(8)))  float  v8f;

// ---------------------------------------------------------------------------
// 1. Graph prep: deg[b,p], normalized adjacency nadj[b,p,q]
// ---------------------------------------------------------------------------
__global__ __launch_bounds__(256) void prep_kernel(const int* __restrict__ adj,
                                                   float* __restrict__ deg,
                                                   float* __restrict__ nadj) {
  __shared__ float dinvS[B_ * P_];
  for (int i = threadIdx.x; i < B_ * P_; i += 256) {
    const int* r = adj + i * P_;
    int s = 0;
    for (int q = 0; q < P_; ++q) s += r[q];
    deg[i] = (float)s;
    dinvS[i] = rsqrtf((float)s + 1.0f);   // Ahat row-sum = deg + 1
  }
  __syncthreads();
  for (int i = threadIdx.x; i < B_ * P_ * P_; i += 256) {
    int b = i / (P_ * P_); int pq = i % (P_ * P_);
    int p = pq / P_, q = pq % P_;
    float ah = (float)adj[i] + (p == q ? 1.0f : 0.0f);
    nadj[i] = dinvS[b * P_ + p] * ah * dinvS[b * P_ + q];
  }
}

// ---------------------------------------------------------------------------
// 2. AtA[p] = A0[p]^T A0[p]  via bf16-split WMMA (C = HtH + HtL + LtH)
//    grid (N/128, N/128, P), block 256 = 8 waves; each block: 128x128 C tile.
//    LDS panels stored transposed [col][m], row stride 40 bf16 (80 B) so each
//    lane's fragment slice is a contiguous, 16B-aligned run -> ds_load_b128.
// ---------------------------------------------------------------------------
#define LSTR 40
__global__ __launch_bounds__(256) void ata_wmma_kernel(const float* __restrict__ A,
                                                       float* __restrict__ AtA) {
  __shared__ __bf16 ldsAh[128][LSTR];
  __shared__ __bf16 ldsAl[128][LSTR];
  __shared__ __bf16 ldsBh[128][LSTR];
  __shared__ __bf16 ldsBl[128][LSTR];

  const int p  = blockIdx.z;
  const int n0 = blockIdx.x * 128;   // output row (n) base
  const int c0 = blockIdx.y * 128;   // output col (k) base
  const int tid  = threadIdx.x;
  const int lane = tid & 31;
  const int wave = tid >> 5;
  const int wr = wave & 3;           // 4 row groups of 32
  const int wc = wave >> 2;          // 2 col groups of 64
  const int half = lane >> 4;
  const int l15  = lane & 15;

  v8f acc[2][4] = {};

  for (int m0 = 0; m0 < M_; m0 += 32) {
    __syncthreads();
    // stage 32 x 128 panels (both operand column ranges), split hi/lo bf16,
    // transposed into LDS. float4 global reads, element LDS writes.
    for (int i = tid; i < 32 * 32; i += 256) {      // 1024 float4 tiles / panel
      int mm = i >> 5;                 // m row in [0,32)
      int c4 = (i & 31) << 2;          // col base in [0,128)
      size_t rowb = ((size_t)p * M_ + (m0 + mm)) * N_;
      float4 va = *(const float4*)&A[rowb + n0 + c4];
      float4 vb = *(const float4*)&A[rowb + c0 + c4];
      float ea[4] = {va.x, va.y, va.z, va.w};
      float eb[4] = {vb.x, vb.y, vb.z, vb.w};
#pragma unroll
      for (int e = 0; e < 4; ++e) {
        __bf16 ha = (__bf16)ea[e];
        ldsAh[c4 + e][mm] = ha;
        ldsAl[c4 + e][mm] = (__bf16)(ea[e] - (float)ha);
        __bf16 hb = (__bf16)eb[e];
        ldsBh[c4 + e][mm] = hb;
        ldsBl[c4 + e][mm] = (__bf16)(eb[e] - (float)hb);
      }
    }
    __syncthreads();

    // prefetch next K-panel while the matrix units chew on this one
    if (m0 + 32 < M_) {
      int mm = tid & 31;
      int seg = (tid >> 5) << 4;       // 8 segments of 16 floats (64B)
      const float* pf = &A[((size_t)p * M_ + (m0 + 32 + mm)) * N_ + n0 + seg];
      __builtin_prefetch(pf, 0, 1);
    }

    // wide fragment loads; A and B both use contiguous k = 16*half + s
    v16bf ahf[2], alf[2], bhf[4], blf[4];
#pragma unroll
    for (int rt = 0; rt < 2; ++rt) {
      int nl = wr * 32 + rt * 16 + l15;
      v8bf h0 = *(const v8bf*)&ldsAh[nl][half * 16];
      v8bf h1 = *(const v8bf*)&ldsAh[nl][half * 16 + 8];
      ahf[rt] = __builtin_shufflevector(h0, h1, 0, 1, 2, 3, 4, 5, 6, 7,
                                        8, 9, 10, 11, 12, 13, 14, 15);
      v8bf l0 = *(const v8bf*)&ldsAl[nl][half * 16];
      v8bf l1 = *(const v8bf*)&ldsAl[nl][half * 16 + 8];
      alf[rt] = __builtin_shufflevector(l0, l1, 0, 1, 2, 3, 4, 5, 6, 7,
                                        8, 9, 10, 11, 12, 13, 14, 15);
    }
#pragma unroll
    for (int ct = 0; ct < 4; ++ct) {
      int cl = wc * 64 + ct * 16 + l15;
      bhf[ct] = *(const v16bf*)&ldsBh[cl][half * 16];
      blf[ct] = *(const v16bf*)&ldsBl[cl][half * 16];
    }
#pragma unroll
    for (int rt = 0; rt < 2; ++rt)
#pragma unroll
      for (int ct = 0; ct < 4; ++ct) {
        acc[rt][ct] = __builtin_amdgcn_wmma_f32_16x16x32_bf16(
            false, ahf[rt], false, bhf[ct], (short)0, acc[rt][ct], false, false);
        acc[rt][ct] = __builtin_amdgcn_wmma_f32_16x16x32_bf16(
            false, ahf[rt], false, blf[ct], (short)0, acc[rt][ct], false, false);
        acc[rt][ct] = __builtin_amdgcn_wmma_f32_16x16x32_bf16(
            false, alf[rt], false, bhf[ct], (short)0, acc[rt][ct], false, false);
      }
  }

  // store: C/D layout — VGPR v: lanes0-15 -> M=v, lanes16-31 -> M=v+8
#pragma unroll
  for (int rt = 0; rt < 2; ++rt)
#pragma unroll
    for (int ct = 0; ct < 4; ++ct)
#pragma unroll
      for (int v = 0; v < 8; ++v) {
        int n = n0 + wr * 32 + rt * 16 + v + half * 8;
        int c = c0 + wc * 64 + ct * 16 + l15;
        AtA[((size_t)p * N_ + n) * N_ + c] = acc[rt][ct][v];
      }
}

// ---------------------------------------------------------------------------
// 3. Atb[b,p,n] = sum_m A[p,m,n] * b[b,p,m]; also fill feat[:,:,N:2N] once
// ---------------------------------------------------------------------------
__global__ __launch_bounds__(256) void atb_kernel(const float* __restrict__ A,
                                                  const float* __restrict__ bvec,
                                                  float* __restrict__ Atb,
                                                  float* __restrict__ feat) {
  __shared__ float bs[M_];
  int p = blockIdx.y, b = blockIdx.z;
  int n = blockIdx.x * 256 + threadIdx.x;
  for (int m = threadIdx.x; m < M_; m += 256)
    bs[m] = bvec[((size_t)b * P_ + p) * M_ + m];
  __syncthreads();
  float acc = 0.f;
  for (int m = 0; m < M_; ++m)
    acc += A[((size_t)p * M_ + m) * N_ + n] * bs[m];
  Atb[((size_t)b * P_ + p) * N_ + n] = acc;
  feat[((size_t)b * P_ + p) * F_ + N_ + n] = acc;
}

// ---------------------------------------------------------------------------
// 4. zero init
// ---------------------------------------------------------------------------
__global__ __launch_bounds__(256) void zero_kernel(float* __restrict__ ptr, int n) {
  int i = blockIdx.x * 256 + threadIdx.x;
  if (i < n) ptr[i] = 0.f;
}

// ---------------------------------------------------------------------------
// 5. AtAy: batched symmetric matvec (AtA[p,n,m] == AtA[p,m,n] -> coalesced)
//    writes feat[:,:,0:N].  grid (N/256, P)
// ---------------------------------------------------------------------------
__global__ __launch_bounds__(256) void ataY_kernel(const float* __restrict__ AtA,
                                                   const float* __restrict__ y,
                                                   float* __restrict__ feat) {
  __shared__ float ys[B_][N_];          // 32 KB
  int p = blockIdx.y;
  int n = blockIdx.x * 256 + threadIdx.x;
  for (int i = threadIdx.x; i < B_ * N_; i += 256) {
    int bb = i >> 10, m = i & (N_ - 1);
    ys[bb][m] = y[((size_t)bb * P_ + p) * N_ + m];
  }
  __syncthreads();
  float acc[B_] = {};
  for (int m = 0; m < N_; ++m) {
    float a = AtA[((size_t)p * N_ + m) * N_ + n];  // symmetric access, coalesced in n
#pragma unroll
    for (int bb = 0; bb < B_; ++bb) acc[bb] += a * ys[bb][m];
  }
#pragma unroll
  for (int bb = 0; bb < B_; ++bb)
    feat[((size_t)bb * P_ + p) * F_ + n] = acc[bb];
}

// ---------------------------------------------------------------------------
// 6. Generic row GEMM: Y[row,:] = X[row,:] @ W (+ bias). block per row.
// ---------------------------------------------------------------------------
__global__ __launch_bounds__(256) void gemm_rows_kernel(const float* __restrict__ X,
                                                        const float* __restrict__ W,
                                                        const float* __restrict__ bias,
                                                        float* __restrict__ Y,
                                                        int fin, int fout) {
  __shared__ float xr[10240];           // max fin (decoder layer 1)
  int row = blockIdx.x;
  const float* xp = X + (size_t)row * fin;
  for (int k = threadIdx.x; k < fin; k += 256) xr[k] = xp[k];
  __syncthreads();
  for (int c = threadIdx.x; c < fout; c += 256) {
    float acc = bias ? bias[c] : 0.f;
    for (int k = 0; k < fin; ++k) acc += xr[k] * W[(size_t)k * fout + c];
    Y[(size_t)row * fout + c] = acc;
  }
}

// ---------------------------------------------------------------------------
// 7. GCN post: x = BN_over_nodes( leaky( nadj @ XW + bias ) ). block per b.
// ---------------------------------------------------------------------------
__global__ __launch_bounds__(256) void gcn_post_kernel(const float* __restrict__ XW,
                                                       const float* __restrict__ nadj,
                                                       const float* __restrict__ bias,
                                                       const float* __restrict__ g,
                                                       const float* __restrict__ bb,
                                                       float* __restrict__ Xout,
                                                       int fout) {
  __shared__ float adjS[P_ * P_];
  int b = blockIdx.x;
  for (int i = threadIdx.x; i < P_ * P_; i += 256) adjS[i] = nadj[b * P_ * P_ + i];
  __syncthreads();
  for (int f = threadIdx.x; f < fout; f += 256) {
    float xc[P_], z[P_];
#pragma unroll
    for (int q = 0; q < P_; ++q)
      xc[q] = XW[((size_t)b * P_ + q) * fout + f];
    float s = 0.f, s2 = 0.f;
#pragma unroll
    for (int p = 0; p < P_; ++p) {
      float a = bias[f];
#pragma unroll
      for (int q = 0; q < P_; ++q) a += adjS[p * P_ + q] * xc[q];
      a = (a > 0.f) ? a : NEGS * a;     // leaky relu
      z[p] = a; s += a; s2 += a * a;
    }
    float mu = s * (1.0f / P_);
    float var = s2 * (1.0f / P_) - mu * mu;
    float inv = rsqrtf(var + EPSV) * g[f];
#pragma unroll
    for (int p = 0; p < P_; ++p)
      Xout[((size_t)b * P_ + p) * fout + f] = (z[p] - mu) * inv + bb[f];
  }
}

// ---------------------------------------------------------------------------
// 8. Row LayerNorm (optional trailing leaky relu). block per row.
// ---------------------------------------------------------------------------
__global__ __launch_bounds__(256) void ln_rows_kernel(float* __restrict__ X,
                                                      const float* __restrict__ g,
                                                      const float* __restrict__ bb,
                                                      int width, int leaky) {
  __shared__ float red[2][256];
  int row = blockIdx.x;
  float* xp = X + (size_t)row * width;
  float s = 0.f, s2 = 0.f;
  for (int i = threadIdx.x; i < width; i += 256) { float v = xp[i]; s += v; s2 += v * v; }
  red[0][threadIdx.x] = s; red[1][threadIdx.x] = s2;
  __syncthreads();
  for (int off = 128; off > 0; off >>= 1) {
    if ((int)threadIdx.x < off) {
      red[0][threadIdx.x] += red[0][threadIdx.x + off];
      red[1][threadIdx.x] += red[1][threadIdx.x + off];
    }
    __syncthreads();
  }
  float mu  = red[0][0] / (float)width;
  float var = red[1][0] / (float)width - mu * mu;
  float inv = rsqrtf(var + EPSV);
  for (int i = threadIdx.x; i < width; i += 256) {
    float v = (xp[i] - mu) * inv * g[i] + bb[i];
    if (leaky) v = (v > 0.f) ? v : NEGS * v;
    xp[i] = v;
  }
}

// ---------------------------------------------------------------------------
// 9. Hyper-parameter head: sigmoid(h @ Wfc + bfc), clip, scale
// ---------------------------------------------------------------------------
__global__ void hyp_kernel(const float* __restrict__ hin,
                           const float* __restrict__ Wfc,
                           const float* __restrict__ bfc,
                           float* __restrict__ hyp) {
  int t = threadIdx.x;
  if (t >= B_ * 4) return;
  int b = t >> 2, j = t & 3;
  float acc = bfc[j];
  for (int k = 0; k < H_; ++k) acc += hin[b * H_ + k] * Wfc[k * 4 + j];
  float s = 1.0f / (1.0f + __expf(-acc));
  s = fminf(fmaxf(s, 1e-6f), 1.0f);
  const float mx = (j == 0) ? 0.1f : 1.0f;
  hyp[b * 4 + j] = fminf(s * mx, mx);
}

// ---------------------------------------------------------------------------
// 10. y update (proximal gradient step); writes Y output slice
// ---------------------------------------------------------------------------
__global__ __launch_bounds__(256) void update_y_kernel(const float* __restrict__ feat,
                                                       const float* __restrict__ Atb,
                                                       float* __restrict__ y,
                                                       const float* __restrict__ U,
                                                       const float* __restrict__ delta,
                                                       const float* __restrict__ deg,
                                                       const float* __restrict__ hyp,
                                                       float* __restrict__ Yout) {
  int idx = blockIdx.x * 256 + threadIdx.x;      // over B*P*N
  int b = idx / (P_ * N_);
  int pn = idx % (P_ * N_);
  int p = pn / N_, n = pn % N_;
  float alpha = hyp[b * 4 + 0], tau = hyp[b * 4 + 1], rho = hyp[b * 4 + 2];
  float atay = feat[((size_t)b * P_ + p) * F_ + n];
  float yv = y[idx];
  float sgn = (yv > 0.f) ? 1.f : ((yv < 0.f) ? -1.f : 0.f);
  float gr = atay - Atb[idx] + sgn * tau + U[idx] * deg[b * P_ + p] + delta[idx] * rho;
  gr = fminf(fmaxf(gr, -10.f), 10.f);
  yv = fminf(fmaxf(yv - alpha * gr, -100.f), 100.f);
  y[idx] = yv;
  Yout[idx] = yv;
}

// ---------------------------------------------------------------------------
// 11. delta / U update from new y
// ---------------------------------------------------------------------------
__global__ __launch_bounds__(256) void update_delta_kernel(const int* __restrict__ adj,
                                                           const float* __restrict__ y,
                                                           const float* __restrict__ deg,
                                                           const float* __restrict__ hyp,
                                                           float* __restrict__ delta,
                                                           float* __restrict__ U) {
  int idx = blockIdx.x * 256 + threadIdx.x;
  int b = idx / (P_ * N_);
  int pn = idx % (P_ * N_);
  int p = pn / N_, n = pn % N_;
  float eta = hyp[b * 4 + 3];
  const int* arow = adj + (b * P_ + p) * P_;
  float s = 0.f;
#pragma unroll
  for (int q = 0; q < P_; ++q)
    s += (float)arow[q] * y[((size_t)b * P_ + q) * N_ + n];
  float d = 2.f * (deg[b * P_ + p] * y[idx] - s);
  d = fminf(fmaxf(d, -10.f), 10.f);
  delta[idx] = d;
  float u = U[idx] + d * eta;
  U[idx] = fminf(fmaxf(u, -100.f), 100.f);
}

// ---------------------------------------------------------------------------
// host-side orchestration
// ---------------------------------------------------------------------------
extern "C" void kernel_launch(void* const* d_in, const int* in_sizes, int n_in,
                              void* d_out, int out_size, void* d_ws, size_t ws_size,
                              hipStream_t stream) {
  (void)in_sizes; (void)n_in; (void)out_size; (void)ws_size;
  const float* bvec = (const float*)d_in[0];
  const int*   adj  = (const int*)d_in[1];
  const float* A    = (const float*)d_in[2];
  // params (insertion order)
  const float* Wc[5]  = {(const float*)d_in[3],  (const float*)d_in[7],
                         (const float*)d_in[11], (const float*)d_in[15],
                         (const float*)d_in[19]};
  const float* bc[5]  = {(const float*)d_in[4],  (const float*)d_in[8],
                         (const float*)d_in[12], (const float*)d_in[16],
                         (const float*)d_in[20]};
  const float* gbn[5] = {(const float*)d_in[5],  (const float*)d_in[9],
                         (const float*)d_in[13], (const float*)d_in[17],
                         (const float*)d_in[21]};
  const float* bbn[5] = {(const float*)d_in[6],  (const float*)d_in[10],
                         (const float*)d_in[14], (const float*)d_in[18],
                         (const float*)d_in[22]};
  const float* g_ln = (const float*)d_in[23];
  const float* b_ln = (const float*)d_in[24];
  const float* Wd[3]  = {(const float*)d_in[25], (const float*)d_in[29],
                         (const float*)d_in[33]};
  const float* bd[3]  = {(const float*)d_in[26], (const float*)d_in[30],
                         (const float*)d_in[34]};
  const float* gld[3] = {(const float*)d_in[27], (const float*)d_in[31],
                         (const float*)d_in[35]};
  const float* bld[3] = {(const float*)d_in[28], (const float*)d_in[32],
                         (const float*)d_in[36]};
  const float* Wfc = (const float*)d_in[37];
  const float* bfc = (const float*)d_in[38];

  float* Yout = (float*)d_out;

  // workspace layout
  float* ws   = (float*)d_ws;
  float* AtA  = ws;                      // 20*1024*1024
  float* Atb  = AtA + (size_t)P_ * N_ * N_;
  float* yv   = Atb + (size_t)B_ * P_ * N_;
  float* Uv   = yv  + (size_t)B_ * P_ * N_;
  float* dv   = Uv  + (size_t)B_ * P_ * N_;
  float* feat = dv  + (size_t)B_ * P_ * N_;          // B*P*2N
  float* W1   = feat + (size_t)B_ * P_ * F_;          // xw scratch (B*P*512)
  float* W0   = W1  + (size_t)B_ * P_ * H4_;          // x  scratch (B*P*512)
  float* hA   = W0  + (size_t)B_ * P_ * H4_;          // B*512
  float* hB   = hA  + (size_t)B_ * H4_;               // B*256
  float* hC   = hB  + (size_t)B_ * 256;               // B*128
  float* deg  = hC  + (size_t)B_ * H_;                // B*P
  float* nadj = deg + B_ * P_;                        // B*P*P
  float* hyp  = nadj + B_ * P_ * P_;                  // B*4

  const int BPN = B_ * P_ * N_;
  const int fout[5] = {H_, 2 * H_, 4 * H_, 4 * H_, 4 * H_};

  prep_kernel<<<1, 256, 0, stream>>>(adj, deg, nadj);
  ata_wmma_kernel<<<dim3(N_ / 128, N_ / 128, P_), 256, 0, stream>>>(A, AtA);
  atb_kernel<<<dim3(N_ / 256, P_, B_), 256, 0, stream>>>(A, bvec, Atb, feat);
  zero_kernel<<<(3 * BPN + 255) / 256, 256, 0, stream>>>(yv, 3 * BPN);  // y, U, delta contiguous

  for (int k = 0; k < K_; ++k) {
    ataY_kernel<<<dim3(N_ / 256, P_), 256, 0, stream>>>(AtA, yv, feat);

    // encoder: 5 GCN layers
    const float* xin = feat;
    int fin = F_;
    for (int i = 0; i < 5; ++i) {
      gemm_rows_kernel<<<B_ * P_, 256, 0, stream>>>(xin, Wc[i], nullptr, W1, fin, fout[i]);
      gcn_post_kernel<<<B_, 256, 0, stream>>>(W1, nadj, bc[i], gbn[i], bbn[i], W0, fout[i]);
      xin = W0;
      fin = fout[i];
    }
    ln_rows_kernel<<<B_ * P_, 256, 0, stream>>>(W0, g_ln, b_ln, H4_, 0);

    // decoder on h = W0 viewed as (B, P*4H)
    gemm_rows_kernel<<<B_, 256, 0, stream>>>(W0, Wd[0], bd[0], hA, P_ * H4_, H4_);
    ln_rows_kernel<<<B_, 256, 0, stream>>>(hA, gld[0], bld[0], H4_, 1);
    gemm_rows_kernel<<<B_, 256, 0, stream>>>(hA, Wd[1], bd[1], hB, H4_, 2 * H_);
    ln_rows_kernel<<<B_, 256, 0, stream>>>(hB, gld[1], bld[1], 2 * H_, 1);
    gemm_rows_kernel<<<B_, 256, 0, stream>>>(hB, Wd[2], bd[2], hC, 2 * H_, H_);
    ln_rows_kernel<<<B_, 256, 0, stream>>>(hC, gld[2], bld[2], H_, 1);

    hyp_kernel<<<1, 32, 0, stream>>>(hC, Wfc, bfc, hyp);

    update_y_kernel<<<BPN / 256, 256, 0, stream>>>(feat, Atb, yv, Uv, dv, deg, hyp,
                                                   Yout + (size_t)k * BPN);
    update_delta_kernel<<<BPN / 256, 256, 0, stream>>>(adj, yv, deg, hyp, dv, Uv);
  }
}
